// CausalSelfAttention_42554535969560
// MI455X (gfx1250) — compile-verified, matrix operand order checked by reading, not run on hardware
//
#include <hip/hip_runtime.h>

typedef __bf16 bf16;
typedef __attribute__((ext_vector_type(16))) __bf16 bf16x16;
typedef __attribute__((ext_vector_type(8)))  __bf16 bf16x8;
typedef __attribute__((ext_vector_type(8)))  float  f32x8;
typedef __attribute__((ext_vector_type(4)))  unsigned u32x4;
typedef __attribute__((ext_vector_type(8)))  unsigned u32x8;

#define D_MODEL  1024
#define N_HEAD   16
#define HEAD_DIM 64
#define SEQ      2048
#define BATCH    2
#define MTOK     (BATCH * SEQ)   // 4096 tokens

// ---------------------------------------------------------------------------
// WMMA helpers (CDNA5 gfx1250, wave32)
// ---------------------------------------------------------------------------
__device__ __forceinline__ f32x8 wmma_bf16(bf16x16 a, bf16x16 b, f32x8 c) {
  return __builtin_amdgcn_wmma_f32_16x16x32_bf16(
      /*neg_a=*/false, a, /*neg_b=*/false, b,
      /*c_mod=*/(short)0, c, /*reuse_a=*/false, /*reuse_b=*/false);
}

// 16x32 bf16 operand fragment (ISA 7.12.2): lanes 0-15 hold K {0..7,16..23},
// lanes 16-31 hold K {8..15,24..31}: two contiguous 16B chunks.
__device__ __forceinline__ bf16x16 load_frag(const bf16* row_k0) {
  const int lane = threadIdx.x & 31;
  const int base = (lane & 16) ? 8 : 0;
  bf16x16 f;
  ((bf16x8*)&f)[0] = *(const bf16x8*)(row_k0 + base);
  ((bf16x8*)&f)[1] = *(const bf16x8*)(row_k0 + base + 16);
  return f;
}

// 16-lane-group reductions (wave32; xor offsets < 16 stay inside each half)
__device__ __forceinline__ float redmax16(float v) {
  v = fmaxf(v, __shfl_xor(v, 8));
  v = fmaxf(v, __shfl_xor(v, 4));
  v = fmaxf(v, __shfl_xor(v, 2));
  v = fmaxf(v, __shfl_xor(v, 1));
  return v;
}
__device__ __forceinline__ float redsum16(float v) {
  v += __shfl_xor(v, 8);
  v += __shfl_xor(v, 4);
  v += __shfl_xor(v, 2);
  v += __shfl_xor(v, 1);
  return v;
}

// LDS byte offset of a shared-memory pointer: generic LDS addresses carry the
// wave-relative LDS byte address in the low 32 bits (ISA 10.2 aperture calc).
__device__ __forceinline__ unsigned lds_off_of(const void* p) {
  return (unsigned)(size_t)p;
}

// ---------------------------------------------------------------------------
// Tensor Data Mover: async DMA of a 2D bf16 tile (tile_rows x tile_elems)
// from global (row stride = stride_elems) into LDS at lds_off.
// D# layout per CDNA5 ISA (2D: groups 0 and 1).
// Wave-level op (EXEC ignored) -> caller must issue from a single wave.
// ---------------------------------------------------------------------------
__device__ __forceinline__ void tdm_load_2d(unsigned lds_off, const bf16* gsrc,
                                            unsigned stride_elems,
                                            unsigned nrows_total,
                                            unsigned tile_rows,
                                            unsigned tile_elems) {
  unsigned long long ga = (unsigned long long)(size_t)gsrc;
  u32x4 g0 = {
      1u,                                             // count=1, user mode
      lds_off,                                        // lds_addr (bytes)
      (unsigned)ga,                                   // global_addr[31:0]
      ((unsigned)(ga >> 32) & 0x01FFFFFFu) | 0x80000000u  // addr[56:32], type=2
  };
  u32x8 g1 = {
      (1u << 16),                                     // data_size=1 (2 bytes)
      (stride_elems & 0xFFFFu) << 16,                 // tensor_dim0[15:0]
      ((stride_elems >> 16) & 0xFFFFu) | ((nrows_total & 0xFFFFu) << 16),
      ((nrows_total >> 16) & 0xFFFFu) | (tile_elems << 16),  // tile_dim0
      tile_rows & 0xFFFFu,                            // tile_dim1 (tile_dim2=0)
      stride_elems,                                   // tensor_dim0_stride[31:0]
      0u, 0u
  };
  asm volatile("tensor_load_to_lds %0, %1" :: "s"(g0), "s"(g1) : "memory");
}

// ---------------------------------------------------------------------------
// Conversion kernels
// ---------------------------------------------------------------------------
__global__ __launch_bounds__(256) void k_cvt_bf16(const float* __restrict__ in,
                                                  bf16* __restrict__ out, int n) {
  int i = blockIdx.x * 256 + threadIdx.x;
  if (i < n) out[i] = (bf16)in[i];
}

// in: [rows][cols] f32 -> out: [cols][rows] bf16 (weight -> W^T)
__global__ __launch_bounds__(256) void k_transpose_bf16(const float* __restrict__ in,
                                                        bf16* __restrict__ out,
                                                        int rows, int cols) {
  int i = blockIdx.x * 256 + threadIdx.x;
  if (i < rows * cols) {
    int r = i / cols, c = i - r * cols;
    out[(size_t)c * rows + r] = (bf16)in[i];
  }
}

// ---------------------------------------------------------------------------
// GEMM core: C[256 x 64] block tile, 8 waves x (32 x 64), K = 1024.
// B slab (64 rows x 64 k) staged in LDS by the TDM, double buffered;
// A fragments double buffered in registers. K-loop fully unrolled so the
// ping-pong index is a compile-time constant (fragments stay in VGPRs).
// ---------------------------------------------------------------------------
__device__ __forceinline__ void gemm_core(const bf16* __restrict__ A,
                                          const bf16* __restrict__ Wt,
                                          unsigned nrows_total,
                                          bf16* b0, bf16* b1,
                                          int m0w, int n0, f32x8 acc[2][4]) {
  const int lane = threadIdx.x & 31;
  const int wave = threadIdx.x >> 5;
  const unsigned loff0 = lds_off_of(b0);
  const unsigned loff1 = lds_off_of(b1);
  const bf16* wbase = Wt + (size_t)n0 * D_MODEL;

  if (wave == 0) tdm_load_2d(loff0, wbase, D_MODEL, nrows_total, 64, 64);

  const bf16* arow0 = A + (size_t)(m0w + (lane & 15)) * D_MODEL;
  const bf16* arow1 = arow0 + (size_t)16 * D_MODEL;

  bf16x16 af[2][2][2];  // [buf][m-subtile][k-step]
  af[0][0][0] = load_frag(arow0);
  af[0][0][1] = load_frag(arow0 + 32);
  af[0][1][0] = load_frag(arow1);
  af[0][1][1] = load_frag(arow1 + 32);

  if (wave == 0) __builtin_amdgcn_s_wait_tensorcnt(0);
  __syncthreads();

#pragma unroll
  for (int k = 0; k < D_MODEL; k += 64) {
    const int cur = (k >> 6) & 1;
    const int nxt = cur ^ 1;
    if (k + 64 < D_MODEL) {
      if (wave == 0)
        tdm_load_2d(cur ? loff0 : loff1, wbase + (k + 64), D_MODEL,
                    nrows_total, 64, 64);
      af[nxt][0][0] = load_frag(arow0 + k + 64);
      af[nxt][0][1] = load_frag(arow0 + k + 96);
      af[nxt][1][0] = load_frag(arow1 + k + 64);
      af[nxt][1][1] = load_frag(arow1 + k + 96);
    }
    const bf16* bb = cur ? b1 : b0;
#pragma unroll
    for (int kk = 0; kk < 2; ++kk) {
#pragma unroll
      for (int t = 0; t < 4; ++t) {
        bf16x16 bfrag = load_frag(bb + (size_t)(t * 16 + (lane & 15)) * 64 + kk * 32);
        acc[0][t] = wmma_bf16(af[cur][0][kk], bfrag, acc[0][t]);
        acc[1][t] = wmma_bf16(af[cur][1][kk], bfrag, acc[1][t]);
      }
    }
    if (k + 64 < D_MODEL) {
      if (wave == 0) __builtin_amdgcn_s_wait_tensorcnt(0);
      __syncthreads();  // next slab visible to all; this slab's reads done
    }
  }
}

// ---------------------------------------------------------------------------
// QKV GEMM: qkv[m][n] = x . w_qkv + b_qkv, scattered to Q/K/V [B][H][T][Dh]
// ---------------------------------------------------------------------------
__global__ __launch_bounds__(256) void k_gemm_qkv(const bf16* __restrict__ A,
                                                  const bf16* __restrict__ Wt,
                                                  const float* __restrict__ bias,
                                                  bf16* __restrict__ Qo,
                                                  bf16* __restrict__ Ko,
                                                  bf16* __restrict__ Vo) {
  __shared__ __align__(16) bf16 bs0[64 * 64];
  __shared__ __align__(16) bf16 bs1[64 * 64];
  const int wave = threadIdx.x >> 5;
  const int lane = threadIdx.x & 31;
  const int m0w = blockIdx.x * 256 + wave * 32;
  const int n0 = blockIdx.y * 64;

  f32x8 acc[2][4] = {};
  gemm_core(A, Wt, 3 * D_MODEL, bs0, bs1, m0w, n0, acc);

  const int hi = (lane >> 4) & 1;
  const int sec = n0 >> 10;                 // 0=Q 1=K 2=V
  const int h = (n0 & (D_MODEL - 1)) >> 6;  // head
  bf16* dst = (sec == 0) ? Qo : (sec == 1) ? Ko : Vo;
#pragma unroll
  for (int mi = 0; mi < 2; ++mi) {
#pragma unroll
    for (int t = 0; t < 4; ++t) {
#pragma unroll
      for (int r = 0; r < 8; ++r) {
        int m = m0w + mi * 16 + r + 8 * hi;   // global token
        int n = n0 + t * 16 + (lane & 15);    // global column in [0,3C)
        int b = m >> 11, tok = m & (SEQ - 1);
        int d = n & (HEAD_DIM - 1);
        float v = acc[mi][t][r] + bias[n];
        dst[(((size_t)(b * N_HEAD + h)) * SEQ + tok) * HEAD_DIM + d] = (bf16)v;
      }
    }
  }
}

// ---------------------------------------------------------------------------
// Projection GEMM: out = y . w_proj + b_proj (f32 out)
// ---------------------------------------------------------------------------
__global__ __launch_bounds__(256) void k_gemm_proj(const bf16* __restrict__ A,
                                                   const bf16* __restrict__ Wt,
                                                   const float* __restrict__ bias,
                                                   float* __restrict__ out) {
  __shared__ __align__(16) bf16 bs0[64 * 64];
  __shared__ __align__(16) bf16 bs1[64 * 64];
  const int wave = threadIdx.x >> 5;
  const int lane = threadIdx.x & 31;
  const int m0w = blockIdx.x * 256 + wave * 32;
  const int n0 = blockIdx.y * 64;

  f32x8 acc[2][4] = {};
  gemm_core(A, Wt, D_MODEL, bs0, bs1, m0w, n0, acc);

  const int hi = (lane >> 4) & 1;
#pragma unroll
  for (int mi = 0; mi < 2; ++mi) {
#pragma unroll
    for (int t = 0; t < 4; ++t) {
#pragma unroll
      for (int r = 0; r < 8; ++r) {
        int m = m0w + mi * 16 + r + 8 * hi;
        int n = n0 + t * 16 + (lane & 15);
        out[(size_t)m * D_MODEL + n] = acc[mi][t][r] + bias[n];
      }
    }
  }
}

// ---------------------------------------------------------------------------
// Flash attention: block = (64-query tile, head, batch); 4 waves x 16 rows.
// K tile -> LDS via async global->LDS copies (ASYNCcnt); V transposed into
// LDS; S = Q K^T and O += P V via WMMA; online softmax in f32; P re-layout
// through per-wave LDS (bf16).
// ---------------------------------------------------------------------------
__global__ __launch_bounds__(128) void k_attn(const bf16* __restrict__ Q,
                                              const bf16* __restrict__ K,
                                              const bf16* __restrict__ V,
                                              bf16* __restrict__ Y) {
  __shared__ __align__(16) bf16 kbuf[64 * HEAD_DIM];     // K tile [key][d]
  __shared__ __align__(16) bf16 vT[HEAD_DIM * 64];       // V^T tile [d][key]
  __shared__ __align__(16) bf16 pbuf[4][16 * 64];        // per-wave P tiles

  const int qt = blockIdx.x;
  const int h  = blockIdx.y;
  const int b  = blockIdx.z;
  const int wave = threadIdx.x >> 5;
  const int lane = threadIdx.x & 31;
  const int hi = (lane >> 4) & 1;
  const float NEG_INF = -__builtin_inff();

  const size_t head_off = ((size_t)(b * N_HEAD + h)) * SEQ * HEAD_DIM;
  const bf16* Qh = Q + head_off;
  const bf16* Kh = K + head_off;
  const bf16* Vh = V + head_off;
  const unsigned kbase = lds_off_of(kbuf);

  const int q0 = qt * 64 + wave * 16;
  const bf16* qrow = Qh + (size_t)(q0 + (lane & 15)) * HEAD_DIM;
  bf16x16 qf0 = load_frag(qrow);
  bf16x16 qf1 = load_frag(qrow + 32);

  f32x8 o[4] = {};
  float m_i[8], l_i[8];
#pragma unroll
  for (int r = 0; r < 8; ++r) { m_i[r] = NEG_INF; l_i[r] = 0.f; }

  for (int j = 0; j <= qt; ++j) {
    const int k0g = j * 64;

    __syncthreads();  // previous iteration's LDS reads done

    // ---- async K tile: 8KB = 512 x 16B chunks over 128 threads ----------
    {
      unsigned long long gk = (unsigned long long)(size_t)(Kh + (size_t)k0g * HEAD_DIM);
#pragma unroll
      for (int c = 0; c < 4; ++c) {
        unsigned chunk = threadIdx.x + c * 128;
        unsigned ldsa = kbase + chunk * 16;
        unsigned long long ga = gk + (unsigned long long)chunk * 16;
        asm volatile("global_load_async_to_lds_b128 %0, %1, off"
                     :: "v"(ldsa), "v"(ga) : "memory");
      }
    }
    // ---- V tile transposed into LDS (overlaps with async K) -------------
    for (int idx = threadIdx.x; idx < 64 * HEAD_DIM; idx += 128) {
      int key = idx >> 6, d = idx & 63;
      vT[d * 64 + key] = Vh[(size_t)(k0g + key) * HEAD_DIM + d];
    }
    asm volatile("s_wait_asynccnt 0x0" ::: "memory");
    __syncthreads();  // (also waits dscnt for the vT stores)

    // ---- S = (Q K^T) * 1/sqrt(Dh) ---------------------------------------
    f32x8 s[4];
#pragma unroll
    for (int t = 0; t < 4; ++t) {
      const bf16* krow = &kbuf[(t * 16 + (lane & 15)) * HEAD_DIM];
      bf16x16 kf0 = load_frag(krow);
      bf16x16 kf1 = load_frag(krow + 32);
      f32x8 acc = {};
      acc = wmma_bf16(qf0, kf0, acc);
      acc = wmma_bf16(qf1, kf1, acc);
      s[t] = acc * 0.125f;  // 1/sqrt(64)
    }

    // ---- online softmax (row = r + 8*hi, cols across 16 lanes) ----------
#pragma unroll
    for (int r = 0; r < 8; ++r) {
      if (j == qt) {
        int query = q0 + r + 8 * hi;
#pragma unroll
        for (int t = 0; t < 4; ++t) {
          int key = k0g + t * 16 + (lane & 15);
          if (key > query) s[t][r] = NEG_INF;
        }
      }
      float lm = fmaxf(fmaxf(s[0][r], s[1][r]), fmaxf(s[2][r], s[3][r]));
      lm = redmax16(lm);
      float mnew = fmaxf(m_i[r], lm);
      float alpha = __expf(m_i[r] - mnew);
      float rs = 0.f;
#pragma unroll
      for (int t = 0; t < 4; ++t) {
        float p = __expf(s[t][r] - mnew);
        rs += p;
        pbuf[wave][(r + 8 * hi) * 64 + t * 16 + (lane & 15)] = (bf16)p;
      }
      rs = redsum16(rs);
      l_i[r] = l_i[r] * alpha + rs;
      m_i[r] = mnew;
#pragma unroll
      for (int t = 0; t < 4; ++t) o[t][r] *= alpha;
    }

    // wave-private LDS round trip: LDS ops are in-order per wave
    asm volatile("s_wait_dscnt 0" ::: "memory");

    // ---- O += P * V -----------------------------------------------------
    const bf16* prow = &pbuf[wave][(lane & 15) * 64];
    bf16x16 pf0 = load_frag(prow);
    bf16x16 pf1 = load_frag(prow + 32);
#pragma unroll
    for (int t = 0; t < 4; ++t) {
      const bf16* vrow = &vT[(t * 16 + (lane & 15)) * 64];
      bf16x16 vf0 = load_frag(vrow);
      bf16x16 vf1 = load_frag(vrow + 32);
      o[t] = wmma_bf16(pf0, vf0, o[t]);
      o[t] = wmma_bf16(pf1, vf1, o[t]);
    }
  }

  // ---- epilogue: Y[b][tok][h*64+d] = O / l  (bf16, [B,T,C]) --------------
#pragma unroll
  for (int t = 0; t < 4; ++t) {
#pragma unroll
    for (int r = 0; r < 8; ++r) {
      int m = r + 8 * hi;
      int d = t * 16 + (lane & 15);
      int qg = q0 + m;
      float rl = 1.f / l_i[r];
      Y[((size_t)b * SEQ + qg) * D_MODEL + h * HEAD_DIM + d] = (bf16)(o[t][r] * rl);
    }
  }
}

// ---------------------------------------------------------------------------
// Host launcher
// ---------------------------------------------------------------------------
extern "C" void kernel_launch(void* const* d_in, const int* in_sizes, int n_in,
                              void* d_out, int out_size, void* d_ws, size_t ws_size,
                              hipStream_t stream) {
  (void)in_sizes; (void)n_in; (void)out_size; (void)ws_size;
  const float* x      = (const float*)d_in[0];  // [2,2048,1024]
  const float* w_qkv  = (const float*)d_in[1];  // [1024,3072]
  const float* b_qkv  = (const float*)d_in[2];  // [3072]
  const float* w_proj = (const float*)d_in[3];  // [1024,1024]
  const float* b_proj = (const float*)d_in[4];  // [1024]
  float* out = (float*)d_out;                   // [2,2048,1024]

  bf16* xb     = (bf16*)d_ws;
  bf16* wqkvT  = xb    + (size_t)MTOK * D_MODEL;            // [3072][1024]
  bf16* wprojT = wqkvT + (size_t)3 * D_MODEL * D_MODEL;     // [1024][1024]
  bf16* Qb     = wprojT + (size_t)D_MODEL * D_MODEL;        // [B][H][T][Dh]
  bf16* Kb     = Qb + (size_t)MTOK * D_MODEL;
  bf16* Vb     = Kb + (size_t)MTOK * D_MODEL;
  bf16* Yb     = Vb + (size_t)MTOK * D_MODEL;               // [B][T][C]

  {
    int n = MTOK * D_MODEL;
    k_cvt_bf16<<<(n + 255) / 256, 256, 0, stream>>>(x, xb, n);
  }
  {
    int n = D_MODEL * 3 * D_MODEL;
    k_transpose_bf16<<<(n + 255) / 256, 256, 0, stream>>>(w_qkv, wqkvT, D_MODEL, 3 * D_MODEL);
  }
  {
    int n = D_MODEL * D_MODEL;
    k_transpose_bf16<<<(n + 255) / 256, 256, 0, stream>>>(w_proj, wprojT, D_MODEL, D_MODEL);
  }

  k_gemm_qkv<<<dim3(MTOK / 256, (3 * D_MODEL) / 64), 256, 0, stream>>>(
      xb, wqkvT, b_qkv, Qb, Kb, Vb);

  k_attn<<<dim3(SEQ / 64, N_HEAD, BATCH), 128, 0, stream>>>(Qb, Kb, Vb, Yb);

  k_gemm_proj<<<dim3(MTOK / 256, D_MODEL / 64), 256, 0, stream>>>(
      Yb, wprojT, b_proj, out);
}
